// MambaMinimal_4827543241055
// MI455X (gfx1250) — matrix operand order repaired and checked by reading, not compiled, not run on hardware
//
#include <hip/hip_runtime.h>
#include <hip/hip_bf16.h>

// ---------------------------------------------------------------------------
// Mamba minimal block for MI455X (gfx1250, wave32, WMMA bf16 compute)
//   D_MODEL=1024, D_STATE=16, D_CONV=4, D_INNER=2048, B=2, L=1024
//
// GEMMs run on v_wmma_f32_16x16x32_bf16 with fp32 accumulators. All GEMM
// operands are pre-converted fp32->bf16 once. K-loop is manually unrolled x2
// with ping-pong fragment buffers: loads for one buffer overlap WMMAs on the
// other, with no register-rotation moves and no WMMA->VALU hazard stalls.
// ---------------------------------------------------------------------------

#define D_MODEL 1024
#define D_STATE 16
#define D_CONV  4
#define D_INNER 2048
#define BATCH   2
#define SEQ     1024
#define ROWS    (BATCH * SEQ)      // 2048 rows of (b,l)

typedef __attribute__((ext_vector_type(16))) __bf16 v16bf;
typedef __attribute__((ext_vector_type(8)))  __bf16 v8bf;
typedef __attribute__((ext_vector_type(8)))  float  v8f;

__device__ __forceinline__ float sigmoid_f(float x) {
    return 1.0f / (1.0f + __expf(-x));
}

#define CAT16(lo, hi) __builtin_shufflevector(lo, hi, \
    0, 1, 2, 3, 4, 5, 6, 7, 8, 9, 10, 11, 12, 13, 14, 15)

// ---------------------------------------------------------------------------
// fp32 -> bf16 streaming conversion, 8 elements / thread.
// ---------------------------------------------------------------------------
__global__ __launch_bounds__(256)
void cvt_f32_bf16(const float* __restrict__ src, __bf16* __restrict__ dst, int n)
{
    const int i = (blockIdx.x * 256 + threadIdx.x) * 8;
    if (i >= n) return;
    float4 a = *(const float4*)(src + i);
    float4 b = *(const float4*)(src + i + 4);
    v8bf o;
    o[0] = (__bf16)a.x; o[1] = (__bf16)a.y; o[2] = (__bf16)a.z; o[3] = (__bf16)a.w;
    o[4] = (__bf16)b.x; o[5] = (__bf16)b.y; o[6] = (__bf16)b.z; o[7] = (__bf16)b.w;
    *(v8bf*)(dst + i) = o;
}

// ---------------------------------------------------------------------------
// Fragment helpers for the WMMA GEMM.
// ---------------------------------------------------------------------------
template <int MT, int NT>
__device__ __forceinline__ void load_frags(
    const __bf16* const (&aP)[MT], const __bf16* const (&bP)[NT], int k,
    v8bf (&aLo)[MT], v8bf (&aHi)[MT], v8bf (&bLo)[NT], v8bf (&bHi)[NT])
{
    #pragma unroll
    for (int i = 0; i < MT; ++i) {
        aLo[i] = *(const v8bf*)(aP[i] + k);
        aHi[i] = *(const v8bf*)(aP[i] + k + 16);
    }
    #pragma unroll
    for (int t = 0; t < NT; ++t) {
        bLo[t] = *(const v8bf*)(bP[t] + k);
        bHi[t] = *(const v8bf*)(bP[t] + k + 16);
    }
}

template <int MT, int NT>
__device__ __forceinline__ void mma_frags(
    const v8bf (&aLo)[MT], const v8bf (&aHi)[MT],
    const v8bf (&bLo)[NT], const v8bf (&bHi)[NT],
    v8f (&acc)[MT][NT])
{
    #pragma unroll
    for (int i = 0; i < MT; ++i) {
        const v16bf af = CAT16(aLo[i], aHi[i]);
        #pragma unroll
        for (int t = 0; t < NT; ++t) {
            const v16bf bf_ = CAT16(bLo[t], bHi[t]);
            acc[i][t] = __builtin_amdgcn_wmma_f32_16x16x32_bf16(
                false, af, false, bf_, (short)0, acc[i][t], false, false);
        }
    }
}

// ---------------------------------------------------------------------------
// WMMA GEMM (NT layout): C[M,N] = A[M,K] * B[N,K]^T, bf16 in, fp32 out.
// Wave tile: MT*16 (M) x NT*16 (N); block = 8 waves stacked in M.
// K must be a multiple of 64. Ping-pong software pipeline (no reg rotation).
// mode 0: C = v        mode 1: C = softplus(v + bias[n])
//
// Fragment layouts (CDNA5 ISA 7.12.2, wave32):
//   A 16x32 bf16: lane row = lane&15; elems 0..7 -> K=kb..kb+7,
//                 elems 8..15 -> K=kb+16..kb+23, kb = (lane>>4)*8
//   B 32x16 bf16: lane col = lane&15; elems 0..7 -> K=kb..kb+7,
//                 elems 8..15 -> K=kb+16..kb+23, kb = (lane>>4)*8
//   C/D 16x16 f32: lane col = lane&15; vgpr r -> row = r + 8*(lane>>4)
// ---------------------------------------------------------------------------
template <int MT, int NT>
__global__ __launch_bounds__(256)
void wmma_gemm_nt(const __bf16* __restrict__ A, const __bf16* __restrict__ Bw,
                  float* __restrict__ C, const float* __restrict__ bias,
                  int M, int N, int K, int mode)
{
    const int lane = threadIdx.x & 31;
    const int wave = threadIdx.x >> 5;                    // 0..7
    const int m0 = blockIdx.y * (8 * MT * 16) + wave * (MT * 16);
    const int n0 = blockIdx.x * (NT * 16);

    const int arow = lane & 15;
    const int akb  = (lane >> 4) << 3;                    // 0 or 8
    const int bcol = lane & 15;
    const int bkb  = (lane >> 4) << 3;                    // 0 or 8

    v8f acc[MT][NT] = {};

    const __bf16* aP[MT];
    const __bf16* bP[NT];
    #pragma unroll
    for (int i = 0; i < MT; ++i)
        aP[i] = A + (size_t)(m0 + i * 16 + arow) * K + akb;
    #pragma unroll
    for (int t = 0; t < NT; ++t)
        bP[t] = Bw + (size_t)(n0 + t * 16 + bcol) * K + bkb;

    v8bf aLo0[MT], aHi0[MT], bLo0[NT], bHi0[NT];   // buffer 0
    v8bf aLo1[MT], aHi1[MT], bLo1[NT], bHi1[NT];   // buffer 1

    load_frags<MT, NT>(aP, bP, 0, aLo0, aHi0, bLo0, bHi0);

    int k = 0;
    for (; k + 64 < K; k += 64) {
        load_frags<MT, NT>(aP, bP, k + 32, aLo1, aHi1, bLo1, bHi1);
        mma_frags<MT, NT>(aLo0, aHi0, bLo0, bHi0, acc);
        load_frags<MT, NT>(aP, bP, k + 64, aLo0, aHi0, bLo0, bHi0);
        mma_frags<MT, NT>(aLo1, aHi1, bLo1, bHi1, acc);
    }
    // Tail pair: k == K-64
    load_frags<MT, NT>(aP, bP, k + 32, aLo1, aHi1, bLo1, bHi1);
    mma_frags<MT, NT>(aLo0, aHi0, bLo0, bHi0, acc);
    mma_frags<MT, NT>(aLo1, aHi1, bLo1, bHi1, acc);

    // Epilogue
    const int rOff = (lane >> 4) << 3;                    // 0 or 8
    const int ccol = lane & 15;
    #pragma unroll
    for (int i = 0; i < MT; ++i) {
        #pragma unroll
        for (int t = 0; t < NT; ++t) {
            const int col = n0 + t * 16 + ccol;
            #pragma unroll
            for (int r = 0; r < 8; ++r) {
                const int row = m0 + i * 16 + r + rOff;
                float v = acc[i][t][r];
                if (mode == 1) {
                    v += bias[col];
                    v = (v > 20.0f) ? v : __logf(1.0f + __expf(v));  // softplus
                }
                C[(size_t)row * N + col] = v;
            }
        }
    }
}

// ---------------------------------------------------------------------------
// Depthwise causal conv1d (kernel=4, left pad 3) + bias + SiLU.
// Reads x_in = xz[:, 0:2048] (row stride 4096); writes fp32 xconv (scan input)
// and bf16 xconv_bf (A-operand of the dt / B / C GEMMs).
// ---------------------------------------------------------------------------
__global__ __launch_bounds__(256)
void conv_silu(const float* __restrict__ xz, const float* __restrict__ Wc,
               const float* __restrict__ bc, float* __restrict__ xconv,
               __bf16* __restrict__ xconv_bf)
{
    const int idx = blockIdx.x * 256 + threadIdx.x;  // over ROWS * D_INNER
    const int d   = idx & (D_INNER - 1);
    const int row = idx >> 11;                       // (b*SEQ + l)
    const int l   = row & (SEQ - 1);

    const float w0 = Wc[d * 4 + 0], w1 = Wc[d * 4 + 1];
    const float w2 = Wc[d * 4 + 2], w3 = Wc[d * 4 + 3];

    float acc = bc[d];
    const float* col = xz + d;
    if (l >= 3) acc += w0 * col[(size_t)(row - 3) * 4096];
    if (l >= 2) acc += w1 * col[(size_t)(row - 2) * 4096];
    if (l >= 1) acc += w2 * col[(size_t)(row - 1) * 4096];
    acc += w3 * col[(size_t)row * 4096];

    const float s = acc * sigmoid_f(acc);            // SiLU
    xconv[idx]    = s;
    xconv_bf[idx] = (__bf16)s;
}

// ---------------------------------------------------------------------------
// Selective scan, sequential over L, parallel over (b, d, n).
// Block = 256 threads = 16 d's x 16 n's. 256 blocks = 2 batches x 128 d-blocks.
// n = tid & 15 -> shfl_xor {1,2,4,8} reduces over N inside each 16-lane half
// of the wave32. Lane n==0 applies the silu(z) gate and stores bf16 ygate.
// ---------------------------------------------------------------------------
__global__ __launch_bounds__(256)
void selective_scan(const float* __restrict__ xconv, const float* __restrict__ dt,
                    const float* __restrict__ Bm, const float* __restrict__ Cm,
                    const float* __restrict__ A_log, const float* __restrict__ xz,
                    __bf16* __restrict__ ygate_bf)
{
    const int t  = threadIdx.x;
    const int n  = t & 15;
    const int dl = t >> 4;                           // 0..15
    const int b  = blockIdx.x >> 7;
    const int d  = ((blockIdx.x & 127) << 4) + dl;

    const float Av = -__expf(A_log[(size_t)d * D_STATE + n]);  // A = -exp(A_log)
    float h = 0.0f;

    for (int l = 0; l < SEQ; ++l) {
        const size_t base = (size_t)(b * SEQ + l);
        const float dtv = dt[base * D_INNER + d];
        const float u   = xconv[base * D_INNER + d];
        const float Bv  = Bm[base * D_STATE + n];
        const float Cv  = Cm[base * D_STATE + n];

        const float dA = __expf(dtv * Av);
        h = dA * h + (dtv * Bv) * u;

        float p = h * Cv;
        p += __shfl_xor(p, 1, 32);
        p += __shfl_xor(p, 2, 32);
        p += __shfl_xor(p, 4, 32);
        p += __shfl_xor(p, 8, 32);

        if (n == 0) {
            const float z = xz[base * 4096 + D_INNER + d];
            ygate_bf[base * D_INNER + d] = (__bf16)(p * (z * sigmoid_f(z)));
        }
    }
}

// ---------------------------------------------------------------------------
// Workspace carving (256B aligned chunks), ~100 MB total.
// ---------------------------------------------------------------------------
static inline char* carve(char*& p, size_t bytes) {
    char* r = p;
    p += (bytes + 255) & ~(size_t)255;
    return r;
}

extern "C" void kernel_launch(void* const* d_in, const int* in_sizes, int n_in,
                              void* d_out, int out_size, void* d_ws, size_t ws_size,
                              hipStream_t stream)
{
    const float* x      = (const float*)d_in[0];
    const float* W_in   = (const float*)d_in[1];
    const float* W_conv = (const float*)d_in[2];
    const float* b_conv = (const float*)d_in[3];
    const float* W_dt   = (const float*)d_in[4];
    const float* b_dt   = (const float*)d_in[5];
    const float* W_B    = (const float*)d_in[6];
    const float* W_C    = (const float*)d_in[7];
    const float* A_log  = (const float*)d_in[8];
    const float* W_out  = (const float*)d_in[9];
    float* out = (float*)d_out;

    char* p = (char*)d_ws;
    float*  xz     = (float*) carve(p, (size_t)ROWS * 4096 * 4);
    float*  xconv  = (float*) carve(p, (size_t)ROWS * D_INNER * 4);
    float*  dt     = (float*) carve(p, (size_t)ROWS * D_INNER * 4);
    float*  Bm     = (float*) carve(p, (size_t)ROWS * D_STATE * 4);
    float*  Cm     = (float*) carve(p, (size_t)ROWS * D_STATE * 4);
    __bf16* x_bf   = (__bf16*)carve(p, (size_t)ROWS * D_MODEL * 2);
    __bf16* Win_bf = (__bf16*)carve(p, (size_t)(2 * D_INNER) * D_MODEL * 2);
    __bf16* Wdt_bf = (__bf16*)carve(p, (size_t)D_INNER * D_INNER * 2);
    __bf16* WB_bf  = (__bf16*)carve(p, (size_t)D_STATE * D_INNER * 2);
    __bf16* WC_bf  = (__bf16*)carve(p, (size_t)D_STATE * D_INNER * 2);
    __bf16* Wout_bf= (__bf16*)carve(p, (size_t)D_MODEL * D_INNER * 2);
    __bf16* xc_bf  = (__bf16*)carve(p, (size_t)ROWS * D_INNER * 2);
    __bf16* yg_bf  = (__bf16*)carve(p, (size_t)ROWS * D_INNER * 2);

    const dim3 blk(256);
    auto cvtGrid = [](size_t n) { return dim3((unsigned)((n / 8 + 255) / 256)); };

    // 0) bf16 pre-conversions (one streaming pass each)
    cvt_f32_bf16<<<cvtGrid((size_t)ROWS * D_MODEL), blk, 0, stream>>>(
        x, x_bf, ROWS * D_MODEL);
    cvt_f32_bf16<<<cvtGrid((size_t)2 * D_INNER * D_MODEL), blk, 0, stream>>>(
        W_in, Win_bf, 2 * D_INNER * D_MODEL);
    cvt_f32_bf16<<<cvtGrid((size_t)D_INNER * D_INNER), blk, 0, stream>>>(
        W_dt, Wdt_bf, D_INNER * D_INNER);
    cvt_f32_bf16<<<cvtGrid((size_t)D_STATE * D_INNER), blk, 0, stream>>>(
        W_B, WB_bf, D_STATE * D_INNER);
    cvt_f32_bf16<<<cvtGrid((size_t)D_STATE * D_INNER), blk, 0, stream>>>(
        W_C, WC_bf, D_STATE * D_INNER);
    cvt_f32_bf16<<<cvtGrid((size_t)D_MODEL * D_INNER), blk, 0, stream>>>(
        W_out, Wout_bf, D_MODEL * D_INNER);

    // 1) xz = x @ W_in^T                      (2048 x 4096 x 1024)
    wmma_gemm_nt<2, 4><<<dim3((2 * D_INNER) / 64, ROWS / 256), blk, 0, stream>>>(
        x_bf, Win_bf, xz, nullptr, ROWS, 2 * D_INNER, D_MODEL, 0);

    // 2) depthwise causal conv + bias + SiLU  (fp32 + bf16 outputs)
    conv_silu<<<(ROWS * D_INNER) / 256, blk, 0, stream>>>(
        xz, W_conv, b_conv, xconv, xc_bf);

    // 3) dt = softplus(xconv @ W_dt^T + b_dt) (2048 x 2048 x 2048)
    wmma_gemm_nt<2, 4><<<dim3(D_INNER / 64, ROWS / 256), blk, 0, stream>>>(
        xc_bf, Wdt_bf, dt, b_dt, ROWS, D_INNER, D_INNER, 1);

    // 4) Bm = xconv @ W_B^T                   (2048 x 16 x 2048)
    wmma_gemm_nt<2, 1><<<dim3(1, ROWS / 256), blk, 0, stream>>>(
        xc_bf, WB_bf, Bm, nullptr, ROWS, D_STATE, D_INNER, 0);

    // 5) Cm = xconv @ W_C^T
    wmma_gemm_nt<2, 1><<<dim3(1, ROWS / 256), blk, 0, stream>>>(
        xc_bf, WC_bf, Cm, nullptr, ROWS, D_STATE, D_INNER, 0);

    // 6) selective scan + silu(z) gate -> bf16 ygate
    selective_scan<<<BATCH * (D_INNER / 16), blk, 0, stream>>>(
        xconv, dt, Bm, Cm, A_log, xz, yg_bf);

    // 7) out = ygate @ W_out^T                (2048 x 1024 x 2048)
    wmma_gemm_nt<2, 4><<<dim3(D_MODEL / 64, ROWS / 256), blk, 0, stream>>>(
        yg_bf, Wout_bf, out, nullptr, ROWS, D_MODEL, D_INNER, 0);
}
